// GraphCaseSizeMoE_70875550319092
// MI455X (gfx1250) — matrix operand order repaired and checked by reading, not compiled, not run on hardware
//
#include <hip/hip_runtime.h>
#include <hip/hip_bf16.h>

#define NN      50000
#define EE      800000
#define GG      8
#define IN_RAW  16
#define X0      4
#define IN_DIM  6
#define HID     256
#define OUTD    6
#define NE      4
#define RH      128
#define PRIOR_S 0.35f
#define LN_EPS  1e-5f

typedef _Float16 half_t;
typedef __attribute__((ext_vector_type(16))) _Float16 v16h;
typedef __attribute__((ext_vector_type(8)))  _Float16 v8h;
typedef __attribute__((ext_vector_type(4)))  _Float16 v4h;
typedef __attribute__((ext_vector_type(8)))  float    v8f;
typedef __attribute__((ext_vector_type(4)))  unsigned int u32x4;
typedef __attribute__((ext_vector_type(8)))  int      i32x8;
typedef __attribute__((ext_vector_type(4)))  int      i32x4;

// ---------------- encoder layer 1 (K=6, scalar) ----------------
__global__ void k_enc1(const float* __restrict__ x, const float* __restrict__ w1,
                       const float* __restrict__ b1, half_t* __restrict__ t) {
  int idx = blockIdx.x * blockDim.x + threadIdx.x;
  if (idx >= NN * HID) return;
  int n = idx >> 8, c = idx & 255;
  float acc = b1[c];
  const float* xr = x + n * IN_RAW + X0;
#pragma unroll
  for (int k = 0; k < IN_DIM; ++k) acc = fmaf(xr[k], w1[k * HID + c], acc);
  t[idx] = (half_t)fmaxf(acc, 0.f);
}

// ------------- weight convert f32[K][256] -> f16 Wt[256][ldWt] -------------
__global__ void k_wt(const float* __restrict__ W, half_t* __restrict__ Wt,
                     int K, int ldWt, int kOff) {
  int idx = blockIdx.x * blockDim.x + threadIdx.x;
  if (idx >= K * HID) return;
  int k = idx >> 8, n = idx & 255;
  Wt[n * ldWt + kOff + k] = (half_t)W[k * HID + n];
}

// ---------------- WMMA GEMM: C[n,0:256] = act(A@Wt^T + bias) ----------------
// A split over K: k<256 from A0, k>=256 from A1 (f16 [N][256] row-major).
// Wt: f16 [256][Ktot] (output-col major, contiguous K). The 64-col weight tile
// for this block is staged into LDS once via the Tensor Data Mover.
__global__ __launch_bounds__(256) void k_gemm(
    const half_t* __restrict__ A0, const half_t* __restrict__ A1,
    const half_t* __restrict__ Wt, const float* __restrict__ bias,
    float* __restrict__ Cf32, half_t* __restrict__ Cf16,
    int n_rows, int Ktot, int doRelu) {
  __shared__ half_t bsh[64 * 512];   // 64 KB weight tile: [64 cols][Ktot]

  const int wave   = threadIdx.x >> 5;
  const int lane   = threadIdx.x & 31;
  const int laneLo = lane & 15;
  const int laneHi = lane >> 4;
  const int rowBase = blockIdx.x * 128 + wave * 16;
  const int colBase = blockIdx.y * 64;

  // LDS byte offset of bsh: for flat addresses in the LDS aperture,
  // addr[31:0] is the LDS offset (ISA aperture mapping).
  half_t* bp0 = &bsh[0];
  const unsigned ldsOff = (unsigned)(unsigned long long)(void*)bp0;

  // ---- TDM: stage Wt[colBase:colBase+64, 0:Ktot] -> LDS (wave 0 issues) ----
  if (wave == 0) {
    unsigned long long ga =
        (unsigned long long)(const void*)(Wt + (size_t)colBase * Ktot);
    // D# group 0: count=1 | lds_addr | global_addr(57b) | type=2
    u32x4 g0 = { 1u, ldsOff, (unsigned)ga,
                 (unsigned)((ga >> 32) & 0x01FFFFFFull) | (2u << 30) };
    // D# group 1: data_size=2B; tensor 2D Ktot x 256; tile Ktot x 64; stride Ktot
    i32x8 g1 = { (1 << 16),                                   // data_size code 1
                 (int)((Ktot & 0xffff) << 16),                // tensor_dim0 lo
                 (int)(((Ktot >> 16) & 0xffff) | (256 << 16)),// dim0 hi | dim1 lo
                 (int)((Ktot & 0xffff) << 16),                // dim1 hi=0 | tile_dim0
                 64,                                          // tile_dim1 = 64 rows
                 Ktot,                                        // dim0_stride lo
                 0, 0 };
    i32x4 gz = { 0, 0, 0, 0 };
#if __clang_major__ >= 23
    i32x8 gz8 = { 0, 0, 0, 0, 0, 0, 0, 0 };
    __builtin_amdgcn_tensor_load_to_lds(g0, g1, gz, gz, gz8, 0);
#else
    __builtin_amdgcn_tensor_load_to_lds(g0, g1, gz, gz, 0);
#endif
    __builtin_amdgcn_s_wait_tensorcnt(0);
  }
  __syncthreads();
  // The TDM wrote bsh through the descriptor (invisible to alias analysis):
  // escape the pointer + memory clobber so the LDS reads below stay real.
  asm volatile("" : : "r"(bp0) : "memory");

  int arow = rowBase + laneLo;
  if (arow >= n_rows) arow = n_rows - 1;   // clamp: keep EXEC all-ones for WMMA

  v8f acc[4] = {v8f{}, v8f{}, v8f{}, v8f{}};

  for (int k0 = 0; k0 < Ktot; k0 += 32) {
    const half_t* Abase = (k0 < HID) ? A0 : A1;
    const int kk = (k0 < HID) ? k0 : (k0 - HID);
    // A fragment: lanes 0-15 hold K=kk..kk+15 (contiguous);
    // lanes 16-31 hold K=kk+16..kk+31.
    const half_t* ap = Abase + (size_t)arow * HID + kk + laneHi * 16;
    v8h a_lo = *(const v8h*)ap;
    v8h a_hi = *(const v8h*)(ap + 8);
    v16h afrag = __builtin_shufflevector(a_lo, a_hi,
        0, 1, 2, 3, 4, 5, 6, 7, 8, 9, 10, 11, 12, 13, 14, 15);
    // B fragment from LDS tile: col = lane&15 (+16*ct), 16 contiguous K per lane.
#pragma unroll
    for (int ct = 0; ct < 4; ++ct) {
      const half_t* bp = bsh + (size_t)(ct * 16 + laneLo) * Ktot + k0 + laneHi * 16;
      v16h bfrag = *(const v16h*)bp;
      acc[ct] = __builtin_amdgcn_wmma_f32_16x16x32_f16(
          false, afrag, false, bfrag, (short)0, acc[ct], false, false);
    }
  }

#pragma unroll
  for (int ct = 0; ct < 4; ++ct) {
    const int n = colBase + ct * 16 + laneLo;
    const float bv = bias[n];
#pragma unroll
    for (int r = 0; r < 8; ++r) {
      const int row = rowBase + r + laneHi * 8;
      float v = acc[ct][r] + bv;
      if (doRelu) v = fmaxf(v, 0.f);
      if (row < n_rows) {
        if (Cf32) Cf32[(size_t)row * HID + n] = v;
        Cf16[(size_t)row * HID + n] = (half_t)v;
      }
    }
  }
}

// ---------------- graph statistics ----------------
__global__ void k_gcnt_nodes(const int* __restrict__ batch, float* __restrict__ npg) {
  int i = blockIdx.x * blockDim.x + threadIdx.x;
  if (i < NN) atomicAdd(&npg[batch[i]], 1.f);
}
__global__ void k_gcnt_edges(const int* __restrict__ src, const int* __restrict__ batch,
                             float* __restrict__ epg) {
  int i = blockIdx.x * blockDim.x + threadIdx.x;
  if (i < EE) atomicAdd(&epg[batch[src[i]]], 1.f);
}
__global__ void k_gfin(const float* __restrict__ npg, const float* __restrict__ epg,
                       float* __restrict__ gfeat, float* __restrict__ gln) {
  if (threadIdx.x != 0 || blockIdx.x != 0) return;
  float ln_[GG], le_[GG];
  float mu0 = 0.f, mu1 = 0.f;
  for (int g = 0; g < GG; ++g) {
    ln_[g] = logf(fmaxf(npg[g], 1.f));
    le_[g] = log1pf(fmaxf(epg[g], 0.f));
    mu0 += ln_[g]; mu1 += le_[g];
  }
  mu0 /= GG; mu1 /= GG;
  float v0 = 0.f, v1 = 0.f, lmin = ln_[0], lmax = ln_[0];
  for (int g = 0; g < GG; ++g) {
    v0 += (ln_[g] - mu0) * (ln_[g] - mu0);
    v1 += (le_[g] - mu1) * (le_[g] - mu1);
    lmin = fminf(lmin, ln_[g]); lmax = fmaxf(lmax, ln_[g]);
  }
  float sd0 = sqrtf(v0 / GG) + 1e-6f, sd1 = sqrtf(v1 / GG) + 1e-6f;
  float rng = lmax - lmin + 1e-6f;
  for (int g = 0; g < GG; ++g) {
    gfeat[g * 2 + 0] = (ln_[g] - mu0) / sd0;
    gfeat[g * 2 + 1] = (le_[g] - mu1) / sd1;
    gln[g] = (ln_[g] - lmin) / rng;
  }
}

// ---------------- router: LN + softmax + top-2 gate, one block per node ----------------
__global__ __launch_bounds__(RH) void k_router(
    const float* __restrict__ h, const int* __restrict__ batch,
    const float* __restrict__ gfeat, const float* __restrict__ gln,
    const float* __restrict__ w1, const float* __restrict__ b1,
    const float* __restrict__ lng, const float* __restrict__ lnb,
    const float* __restrict__ w2, const float* __restrict__ b2,
    const float* __restrict__ centers, float* __restrict__ sw) {
  __shared__ float sbuf[RH];
  __shared__ float logit[NE];
  const int node = blockIdx.x;
  const int j = threadIdx.x;
  const int g = batch[node];
  const float nf0 = gfeat[g * 2 + 0], nf1 = gfeat[g * 2 + 1], nln = gln[g];
  const float* hr = h + (size_t)node * HID;

  float r = b1[j];
  for (int k = 0; k < HID; ++k) r = fmaf(hr[k], w1[k * RH + j], r);
  r = fmaf(nf0, w1[HID * RH + j], r);
  r = fmaf(nf1, w1[(HID + 1) * RH + j], r);

  sbuf[j] = r; __syncthreads();
  for (int s = RH / 2; s > 0; s >>= 1) { if (j < s) sbuf[j] += sbuf[j + s]; __syncthreads(); }
  const float mu = sbuf[0] * (1.f / RH); __syncthreads();
  const float d = r - mu;
  sbuf[j] = d * d; __syncthreads();
  for (int s = RH / 2; s > 0; s >>= 1) { if (j < s) sbuf[j] += sbuf[j + s]; __syncthreads(); }
  const float var = sbuf[0] * (1.f / RH); __syncthreads();
  const float rr = fmaxf(d * rsqrtf(var + LN_EPS) * lng[j] + lnb[j], 0.f);

  for (int e = 0; e < NE; ++e) {
    sbuf[j] = rr * w2[j * NE + e]; __syncthreads();
    for (int s = RH / 2; s > 0; s >>= 1) { if (j < s) sbuf[j] += sbuf[j + s]; __syncthreads(); }
    if (j == 0) logit[e] = sbuf[0] + b2[e];
    __syncthreads();
  }

  if (j == 0) {
    float lg[NE], p[NE], mx = -1e30f;
    for (int e = 0; e < NE; ++e) {
      float pr = nln - centers[e];
      lg[e] = (1.f - PRIOR_S) * logit[e] + PRIOR_S * (-pr * pr);
      mx = fmaxf(mx, lg[e]);
    }
    float se = 0.f;
    for (int e = 0; e < NE; ++e) { p[e] = expf(lg[e] - mx); se += p[e]; }
    for (int e = 0; e < NE; ++e) p[e] /= se;
    int i0 = 0;
    for (int e = 1; e < NE; ++e) if (p[e] > p[i0]) i0 = e;
    int i1 = -1;
    for (int e = 0; e < NE; ++e) if (e != i0 && (i1 < 0 || p[e] > p[i1])) i1 = e;
    const float s2 = p[i0] + p[i1] + 1e-8f;
    float ov[NE] = {0.f, 0.f, 0.f, 0.f};
    ov[i0] = p[i0] / s2; ov[i1] = p[i1] / s2;
    for (int e = 0; e < NE; ++e) sw[node * NE + e] = ov[e];
  }
}

// ---------------- edge scatter: agg[dst] += val[src] ----------------
__global__ void k_scat_f32(const float* __restrict__ val, const int* __restrict__ src,
                           const int* __restrict__ dst, float* __restrict__ agg) {
  int idx = blockIdx.x * blockDim.x + threadIdx.x;
  if (idx >= EE * 64) return;
  const int e = idx >> 6, c = (idx & 63) << 2;
  const int s = src[e], d = dst[e];
  const float4 v = *(const float4*)(val + (size_t)s * HID + c);
  float* o = agg + (size_t)d * HID + c;
  atomicAdd(o + 0, v.x); atomicAdd(o + 1, v.y);
  atomicAdd(o + 2, v.z); atomicAdd(o + 3, v.w);
}
__global__ void k_scat_f16(const half_t* __restrict__ val, const int* __restrict__ src,
                           const int* __restrict__ dst, float* __restrict__ agg) {
  int idx = blockIdx.x * blockDim.x + threadIdx.x;
  if (idx >= EE * 64) return;
  const int e = idx >> 6, c = (idx & 63) << 2;
  const int s = src[e], d = dst[e];
  const v4h v = *(const v4h*)(val + (size_t)s * HID + c);
  float* o = agg + (size_t)d * HID + c;
  atomicAdd(o + 0, (float)v[0]); atomicAdd(o + 1, (float)v[1]);
  atomicAdd(o + 2, (float)v[2]); atomicAdd(o + 3, (float)v[3]);
}

// ---------------- f32 -> f16 ----------------
__global__ void k_f2h(const float* __restrict__ in, half_t* __restrict__ outp, int n) {
  int i = blockIdx.x * blockDim.x + threadIdx.x;
  if (i < n) outp[i] = (half_t)in[i];
}

// ---------------- expert output epilogue ----------------
__global__ void k_out(const float* __restrict__ agg, const half_t* __restrict__ xe,
                      const float* __restrict__ wro, const float* __restrict__ bro,
                      const float* __restrict__ wrt, const float* __restrict__ sw,
                      float* __restrict__ outp, int expert) {
  int n = blockIdx.x * blockDim.x + threadIdx.x;
  if (n >= NN) return;
  const float s = sw[n * NE + expert];
  float acc[OUTD];
#pragma unroll
  for (int o = 0; o < OUTD; ++o) acc[o] = bro[o];
  const float* ar = agg + (size_t)n * HID;
  const half_t* xr = xe + (size_t)n * HID;
  for (int k = 0; k < HID; ++k) {
    const float a = ar[k];
    const float xv = (float)xr[k];
#pragma unroll
    for (int o = 0; o < OUTD; ++o)
      acc[o] += a * wro[k * OUTD + o] + xv * wrt[k * OUTD + o];
  }
#pragma unroll
  for (int o = 0; o < OUTD; ++o) outp[n * OUTD + o] += s * acc[o];
}

// ======================= host launcher =======================
extern "C" void kernel_launch(void* const* d_in, const int* in_sizes, int n_in,
                              void* d_out, int out_size, void* d_ws, size_t ws_size,
                              hipStream_t stream) {
  const float* x        = (const float*)d_in[0];
  const int*   ei       = (const int*)d_in[1];
  const int*   batch    = (const int*)d_in[2];
  const float* enc_w1   = (const float*)d_in[4];
  const float* enc_b1   = (const float*)d_in[5];
  const float* enc_w2   = (const float*)d_in[6];
  const float* enc_b2   = (const float*)d_in[7];
  const float* rt_w1    = (const float*)d_in[8];
  const float* rt_b1    = (const float*)d_in[9];
  const float* ln_g     = (const float*)d_in[10];
  const float* ln_b     = (const float*)d_in[11];
  const float* rt_w2    = (const float*)d_in[12];
  const float* rt_b2    = (const float*)d_in[13];
  const float* centers  = (const float*)d_in[14];
  const float* w_rel    = (const float*)d_in[15];
  const float* b_rel    = (const float*)d_in[16];
  const float* w_root   = (const float*)d_in[17];
  const float* w_relo   = (const float*)d_in[18];
  const float* b_relo   = (const float*)d_in[19];
  const float* w_rooto  = (const float*)d_in[20];
  float* out = (float*)d_out;

  const int* src = ei;
  const int* dst = ei + EE;

  char* ws = (char*)d_ws;
  size_t off = 0;
  auto alloc = [&](size_t bytes) {
    void* p = ws + off;
    off += (bytes + 255) & ~(size_t)255;
    return p;
  };
  float*  h32    = (float*)alloc((size_t)NN * HID * 4);
  float*  agg32  = (float*)alloc((size_t)NN * HID * 4);
  half_t* h16    = (half_t*)alloc((size_t)NN * HID * 2);
  half_t* aggh16 = (half_t*)alloc((size_t)NN * HID * 2);
  half_t* agg16  = (half_t*)alloc((size_t)NN * HID * 2);
  half_t* xe16   = (half_t*)alloc((size_t)NN * HID * 2);   // also encoder L1 temp
  half_t* xe216  = (half_t*)alloc((size_t)NN * HID * 2);
  float*  sw     = (float*)alloc((size_t)NN * NE * 4);
  float*  gstat  = (float*)alloc(64 * 4);
  half_t* wt     = (half_t*)alloc((size_t)HID * 512 * 2);
  float*  npg = gstat, *epg = gstat + 8, *gfeat = gstat + 16, *gln = gstat + 32;

  const int NH = NN * HID;                 // 12.8M elements
  const dim3 gemmGrid((NN + 127) / 128, HID / 64);
  const int scatBlocks = (EE * 64 + 255) / 256;

  hipMemsetAsync(gstat, 0, 64 * 4, stream);
  hipMemsetAsync(agg32, 0, (size_t)NN * HID * 4, stream);
  hipMemsetAsync(out, 0, (size_t)out_size * 4, stream);

  // encoder layer 1 -> xe16 (temp), then WMMA GEMM layer 2 -> h32/h16
  k_enc1<<<(NH + 255) / 256, 256, 0, stream>>>(x, enc_w1, enc_b1, xe16);
  k_wt<<<(HID * HID + 255) / 256, 256, 0, stream>>>(enc_w2, wt, HID, HID, 0);
  k_gemm<<<gemmGrid, 256, 0, stream>>>(xe16, (const half_t*)nullptr, wt, enc_b2,
                                       h32, h16, NN, HID, 0);

  // graph stats + router
  k_gcnt_nodes<<<(NN + 255) / 256, 256, 0, stream>>>(batch, npg);
  k_gcnt_edges<<<(EE + 255) / 256, 256, 0, stream>>>(src, batch, epg);
  k_gfin<<<1, 32, 0, stream>>>(npg, epg, gfeat, gln);
  k_router<<<NN, RH, 0, stream>>>(h32, batch, gfeat, gln, rt_w1, rt_b1,
                                  ln_g, ln_b, rt_w2, rt_b2, centers, sw);

  // agg_h = segsum(h[src] -> dst), then f16 copy
  k_scat_f32<<<scatBlocks, 256, 0, stream>>>(h32, src, dst, agg32);
  k_f2h<<<(NH + 255) / 256, 256, 0, stream>>>(agg32, aggh16, NH);

  for (int e = 0; e < NE; ++e) {
    const float* wr0 = w_rel  + (size_t)(e * 2 + 0) * HID * HID;
    const float* wt0 = w_root + (size_t)(e * 2 + 0) * HID * HID;
    const float* wr1 = w_rel  + (size_t)(e * 2 + 1) * HID * HID;
    const float* wt1 = w_root + (size_t)(e * 2 + 1) * HID * HID;
    const float* b0  = b_rel  + (size_t)(e * 2 + 0) * HID;
    const float* b1e = b_rel  + (size_t)(e * 2 + 1) * HID;

    // layer 1: xe = relu([agg_h | h] @ [w_rel0; w_root0] + b0)
    k_wt<<<(HID * HID + 255) / 256, 256, 0, stream>>>(wr0, wt, HID, 512, 0);
    k_wt<<<(HID * HID + 255) / 256, 256, 0, stream>>>(wt0, wt, HID, 512, HID);
    k_gemm<<<gemmGrid, 256, 0, stream>>>(aggh16, h16, wt, b0,
                                         (float*)nullptr, xe16, NN, 512, 1);

    hipMemsetAsync(agg32, 0, (size_t)NN * HID * 4, stream);
    k_scat_f16<<<scatBlocks, 256, 0, stream>>>(xe16, src, dst, agg32);
    k_f2h<<<(NH + 255) / 256, 256, 0, stream>>>(agg32, agg16, NH);

    // layer 2: xe2 = relu([agg | xe] @ [w_rel1; w_root1] + b1)
    k_wt<<<(HID * HID + 255) / 256, 256, 0, stream>>>(wr1, wt, HID, 512, 0);
    k_wt<<<(HID * HID + 255) / 256, 256, 0, stream>>>(wt1, wt, HID, 512, HID);
    k_gemm<<<gemmGrid, 256, 0, stream>>>(agg16, xe16, wt, b1e,
                                         (float*)nullptr, xe216, NN, 512, 1);

    hipMemsetAsync(agg32, 0, (size_t)NN * HID * 4, stream);
    k_scat_f16<<<scatBlocks, 256, 0, stream>>>(xe216, src, dst, agg32);

    // ye = agg2 @ w_rel_out + b + xe2 @ w_root_out ; out += sw[:,e] * ye
    k_out<<<(NN + 255) / 256, 256, 0, stream>>>(
        agg32, xe216,
        w_relo + (size_t)e * HID * OUTD, b_relo + (size_t)e * OUTD,
        w_rooto + (size_t)e * HID * OUTD, sw, out, e);
  }
}
